// ScoreBranch_8555574854422
// MI455X (gfx1250) — matrix operand order repaired and checked by reading, not compiled, not run on hardware
//
#include <hip/hip_runtime.h>
#include <hip/hip_bf16.h>

// ---------------------------------------------------------------------------
// Problem constants (from the reference)
// ---------------------------------------------------------------------------
#define BATCH   128
#define C_IN    256
#define C_HID   1024
#define HWDIM   17
#define NPOS    (HWDIM * HWDIM)      // 289 spatial positions
#define MBLK    32                   // positions per workgroup (2 WMMA m-tiles)
#define MBLOCKS 10                   // ceil(289 / 32)
#define KTOT    (C_IN * 9)           // 2304 = im2col K
#define KTILES  (KTOT / 32)          // 72 k-tiles of 32
#define KCHUNK  24                   // k-tiles staged in LDS at a time (3 chunks)
#define NTILES  (C_HID / 16)         // 64 n-tiles of 16

typedef __attribute__((ext_vector_type(16))) __bf16 v16bf;
typedef __attribute__((ext_vector_type(8)))  float  v8f;
typedef __attribute__((ext_vector_type(4)))  unsigned int u32x4;
typedef __attribute__((ext_vector_type(8)))  int          i32x8;
typedef __attribute__((ext_vector_type(4)))  int          i32x4;

__device__ __forceinline__ unsigned short f2bf(float f) {
    // round-to-nearest-even f32 -> bf16
    unsigned u = __float_as_uint(f);
    unsigned r = u + 0x7FFFu + ((u >> 16) & 1u);
    return (unsigned short)(r >> 16);
}

// ---------------------------------------------------------------------------
// Kernel 1: pack w_rnn [1024][256][3][3] f32 into bf16 WMMA-B fragments.
// Frag f = (kt*64 + nt)*32 + lane, 32 bytes (16 bf16) per lane-frag.
// B-fragment element i of lane L: N = nt*16 + (L&15), K = kt*32 + (L>>4)*16 + i.
// K maps to (tap, c) via K = tap*256 + c, tap = dy*3 + dx.
// ---------------------------------------------------------------------------
__global__ __launch_bounds__(256) void pack_w_kernel(const float* __restrict__ w,
                                                     unsigned char* __restrict__ Bp) {
    int f = blockIdx.x * 256 + threadIdx.x;          // [0, 72*64*32)
    if (f >= KTILES * NTILES * 32) return;
    int kt   = f >> 11;
    int r    = f & 2047;
    int nt   = r >> 5;
    int lane = r & 31;
    int N    = nt * 16 + (lane & 15);
    int kb   = (lane >> 4) * 16;

    union { unsigned short s[16]; uint4 u[2]; } frag;
#pragma unroll
    for (int i = 0; i < 16; ++i) {
        int Kg  = kt * 32 + kb + i;
        int tap = Kg >> 8;
        int c   = Kg & 255;
        frag.s[i] = f2bf(w[(N * C_IN + c) * 9 + tap]);
    }
    uint4* o = (uint4*)(Bp + (size_t)f * 32);
    o[0] = frag.u[0];
    o[1] = frag.u[1];
}

// ---------------------------------------------------------------------------
// Kernel 2: fused conv3x3 (implicit GEMM via bf16 WMMA) + bias + tanh + relu
//           + 1x1 conv reduction + bias + sigmoid.
// Grid: (10 m-blocks, 128 batches), 256 threads = 8 waves.
// M-block = 32 rows (2 m-tiles); wave owns 8 n-tiles processed as 4 pairs.
// Inner step: 2 A frags (LDS) + 2 B frags (global) -> 4 WMMAs
//   => 1 ds_load_b128 + 1 global_load_b128 per v_wmma (2x less traffic than
//      the naive 16-row blocking; B L2 reads drop to ~5.9 GB total).
// Accumulators: 4 pairs x 2 nt x 2 mt = 16 x v8f = 128 VGPRs (persist over K).
// ---------------------------------------------------------------------------
__global__ __launch_bounds__(256, 1) void conv_wmma_kernel(
        const float* __restrict__ x,
        const float* __restrict__ b_rnn,
        const float* __restrict__ w_out,
        const float* __restrict__ b_out,
        const unsigned char* __restrict__ Bp,
        float* __restrict__ score) {

    __shared__ __align__(16) unsigned short sA[KCHUNK * 2 * 32 * 16]; // 48 KB
    __shared__ float sS[MBLK];                                        // per-row partial s

    const int mb   = blockIdx.x;
    const int n    = blockIdx.y;
    const int tid  = threadIdx.x;
    const int wv   = tid >> 5;        // wave 0..7
    const int lane = tid & 31;

    if (tid < MBLK) sS[tid] = 0.0f;

    // acc[((p*2 + t)*2) + mt] : pair p, nt sub-index t, m-tile mt
    v8f acc[16];
    v8f zero = {0.f, 0.f, 0.f, 0.f, 0.f, 0.f, 0.f, 0.f};
#pragma unroll
    for (int q = 0; q < 16; ++q) acc[q] = zero;

    for (int kc = 0; kc < 3; ++kc) {
        __syncthreads();
        // ---- build A fragments for this K-chunk (im2col + f32->bf16) ----
        // A-fragment element i of lane L: row M = L&15,
        //   K = kb + i,  kb = (L>>4)*16   (contiguous, matches packed B)
        for (int e = tid; e < KCHUNK * 2 * 512; e += 256) {
            int ktl = e >> 10;            // / (2*512)
            int r   = e & 1023;
            int mt  = r >> 9;
            int rr  = r & 511;
            int fl  = rr >> 4;            // fragment lane
            int i   = rr & 15;            // fragment element
            int row = mt * 16 + (fl & 15);
            int kb  = (fl >> 4) * 16;
            int kk  = kb + i;
            int Kg  = (kc * KCHUNK + ktl) * 32 + kk;
            int tap = Kg >> 8;
            int c   = Kg & 255;
            int m   = mb * MBLK + row;
            int y   = m / HWDIM, xc = m % HWDIM;
            int yy  = y + tap / 3 - 1;
            int xx  = xc + tap % 3 - 1;
            float v = 0.0f;
            if (m < NPOS && (unsigned)yy < (unsigned)HWDIM && (unsigned)xx < (unsigned)HWDIM)
                v = x[((n * C_IN + c) * HWDIM + yy) * HWDIM + xx];
            sA[((ktl * 2 + mt) * 32 + fl) * 16 + i] = f2bf(v);
        }
        __syncthreads();

        // prefetch start of next K-chunk's B stream for this wave (L2 warmup)
        if (kc < 2) {
            size_t nb = ((size_t)(((kc + 1) * KCHUNK) * NTILES + wv * 8) * 32 + lane) * 32;
            __builtin_prefetch((const void*)(Bp + nb), 0, 1);
        }

        // ---- WMMA sweep: 4 nt-pairs per wave, 24 k-tiles this chunk ----
#pragma unroll
        for (int p = 0; p < 4; ++p) {
            int nt0 = wv * 8 + p * 2;
            v8f a00 = acc[p * 4 + 0];     // (nt0,   mt0)
            v8f a01 = acc[p * 4 + 1];     // (nt0,   mt1)
            v8f a10 = acc[p * 4 + 2];     // (nt0+1, mt0)
            v8f a11 = acc[p * 4 + 3];     // (nt0+1, mt1)
            for (int ktl = 0; ktl < KCHUNK; ++ktl) {
                int kt = kc * KCHUNK + ktl;
                union { uint4 u[2]; v16bf v; } af0, af1, bf0, bf1;
                const uint4* ap0 = (const uint4*)&sA[((ktl * 2 + 0) * 32 + lane) * 16];
                const uint4* ap1 = (const uint4*)&sA[((ktl * 2 + 1) * 32 + lane) * 16];
                af0.u[0] = ap0[0]; af0.u[1] = ap0[1];
                af1.u[0] = ap1[0]; af1.u[1] = ap1[1];
                const uint4* bp0 =
                    (const uint4*)(Bp + ((size_t)((kt * NTILES + nt0) * 32 + lane)) * 32);
                const uint4* bp1 =
                    (const uint4*)(Bp + ((size_t)((kt * NTILES + nt0 + 1) * 32 + lane)) * 32);
                bf0.u[0] = bp0[0]; bf0.u[1] = bp0[1];
                bf1.u[0] = bp1[0]; bf1.u[1] = bp1[1];
                a00 = __builtin_amdgcn_wmma_f32_16x16x32_bf16(false, af0.v, false, bf0.v,
                                                              (short)0, a00, false, false);
                a01 = __builtin_amdgcn_wmma_f32_16x16x32_bf16(false, af1.v, false, bf0.v,
                                                              (short)0, a01, false, false);
                a10 = __builtin_amdgcn_wmma_f32_16x16x32_bf16(false, af0.v, false, bf1.v,
                                                              (short)0, a10, false, false);
                a11 = __builtin_amdgcn_wmma_f32_16x16x32_bf16(false, af1.v, false, bf1.v,
                                                              (short)0, a11, false, false);
            }
            acc[p * 4 + 0] = a00;
            acc[p * 4 + 1] = a01;
            acc[p * 4 + 2] = a10;
            acc[p * 4 + 3] = a11;
        }
    }

    // ---- fused epilogue: bias + tanh + relu + (h · w_out) partial sums ----
    // C/D layout: element j of lane L -> M = mt*16 + j + 8*(L>>4),
    //             N = nt*16 + (L&15)
    float partial[2][8];
#pragma unroll
    for (int mt = 0; mt < 2; ++mt)
#pragma unroll
        for (int j = 0; j < 8; ++j) partial[mt][j] = 0.0f;

#pragma unroll
    for (int p = 0; p < 4; ++p) {
#pragma unroll
        for (int t = 0; t < 2; ++t) {
            int nOut = (wv * 8 + p * 2 + t) * 16 + (lane & 15);
            float bb = b_rnn[nOut];
            float wo = w_out[nOut];
#pragma unroll
            for (int mt = 0; mt < 2; ++mt) {
                v8f a = acc[p * 4 + t * 2 + mt];
#pragma unroll
                for (int j = 0; j < 8; ++j) {
                    float h = a[j] + bb;
                    h = tanhf(h);
                    h = fmaxf(h, 0.0f);
                    partial[mt][j] += h * wo;
                }
            }
        }
    }
    int rb = (lane >> 4) * 8;
#pragma unroll
    for (int mt = 0; mt < 2; ++mt)
#pragma unroll
        for (int j = 0; j < 8; ++j)
            atomicAdd(&sS[mt * 16 + rb + j], partial[mt][j]);
    __syncthreads();

    if (tid < MBLK) {
        int m = mb * MBLK + tid;
        if (m < NPOS) {
            float s = sS[tid] + b_out[0];
            score[n * NPOS + m] = 1.0f / (1.0f + expf(-s));
        }
    }
}

// ---------------------------------------------------------------------------
// Kernel 3: per-batch argmax (first occurrence) -> (y, x) as floats.
// ---------------------------------------------------------------------------
__global__ __launch_bounds__(256) void argmax_kernel(const float* __restrict__ score,
                                                     float* __restrict__ pos) {
    __shared__ float sv[256];
    __shared__ int   si[256];
    int n   = blockIdx.x;
    int tid = threadIdx.x;
    const float* s = score + n * NPOS;

    float best = -3.402823466e+38f;
    int   bidx = 0x7FFFFFFF;
    for (int i = tid; i < NPOS; i += 256) {
        float v = s[i];
        if (v > best || (v == best && i < bidx)) { best = v; bidx = i; }
    }
    sv[tid] = best;
    si[tid] = bidx;
    __syncthreads();
    for (int off = 128; off > 0; off >>= 1) {
        if (tid < off) {
            float v2 = sv[tid + off];
            int   i2 = si[tid + off];
            if (v2 > sv[tid] || (v2 == sv[tid] && i2 < si[tid])) {
                sv[tid] = v2;
                si[tid] = i2;
            }
        }
        __syncthreads();
    }
    if (tid == 0) {
        pos[n * 2 + 0] = (float)(si[0] / HWDIM);   // y
        pos[n * 2 + 1] = (float)(si[0] % HWDIM);   // x
    }
}

// ---------------------------------------------------------------------------
// TDM probe (not launched): exercises the CDNA5 Tensor Data Mover path with a
// hand-built D# (ISA 08_async_tensor §8.3/§8.4) and s_wait_tensorcnt.
// This toolchain exposes the 6-arg (clang-23 / therock) builtin:
//   (u32x4 g0, i32x8 g1, i32x4 g2, i32x4 g3, i32x8 g4, i32 cpol)
// Group0: count=1, lds_addr[63:32], global_addr[120:64], type=2 [127:126].
// Group1: data_size=4B, 2-D tile (tile_dim0=512 elem, tile_dim1=1);
// Groups 2/3(/4) zero: unused for a 2-D, non-gather tile.
// ---------------------------------------------------------------------------
__global__ void tdm_probe_kernel(const float* __restrict__ src, float* __restrict__ out) {
    __shared__ __align__(16) float tbuf[512];

    unsigned long long ga = (unsigned long long)(const void*)src;
    unsigned lds = (unsigned)(unsigned long long)(void*)tbuf;

    u32x4 g0;
    g0.x = 1u;                                   // count=1, is_restore=0
    g0.y = lds;                                  // lds_addr (bytes)
    g0.z = (unsigned)ga;                         // global_addr[31:0]
    g0.w = ((unsigned)(ga >> 32) & 0x01FFFFFFu)  // global_addr[56:32]
         | (2u << 30);                           // type=2 ("image")

    i32x8 g1;
    g1[0] = (int)(2u << 16);                     // wg_mask=0, data_size=4B
    g1[1] = (int)((512u & 0xFFFFu) << 16);       // tensor_dim0[15:0] -> bits[63:48]
    g1[2] = (int)(1u << 16);                     // tensor_dim0[31:16]=0, tensor_dim1=1
    g1[3] = (int)(512u << 16);                   // tile_dim0 = 512 elements
    g1[4] = 1;                                   // tile_dim1 = 1, tile_dim2 = 0
    g1[5] = 512;                                 // tensor_dim0_stride[31:0]
    g1[6] = 0;                                   // strides hi
    g1[7] = 0;

    i32x4 g2 = {0, 0, 0, 0};
    i32x4 g3 = {0, 0, 0, 0};
    i32x8 g4 = {0, 0, 0, 0, 0, 0, 0, 0};

    __builtin_amdgcn_tensor_load_to_lds(g0, g1, g2, g3, g4, 0);
    __builtin_amdgcn_s_wait_tensorcnt(0);
    __syncthreads();
    out[threadIdx.x] = tbuf[threadIdx.x];
}

// ---------------------------------------------------------------------------
extern "C" void kernel_launch(void* const* d_in, const int* in_sizes, int n_in,
                              void* d_out, int out_size, void* d_ws, size_t ws_size,
                              hipStream_t stream) {
    const float* x     = (const float*)d_in[0];   // [128,256,17,17]
    const float* w_rnn = (const float*)d_in[1];   // [1024,256,3,3]
    const float* b_rnn = (const float*)d_in[2];   // [1024]
    const float* w_out = (const float*)d_in[3];   // [1,1024,1,1]
    const float* b_out = (const float*)d_in[4];   // [1]

    float* score = (float*)d_out;                 // [128,1,17,17] = 36992 floats
    float* pos   = score + BATCH * NPOS;          // [128,2]

    unsigned char* Bp = (unsigned char*)d_ws;     // 4.72 MB packed bf16 B fragments

    // 1) pack weights into WMMA-B fragment layout (recomputed each call: deterministic)
    int nfrag = KTILES * NTILES * 32;
    pack_w_kernel<<<(nfrag + 255) / 256, 256, 0, stream>>>(w_rnn, Bp);

    // 2) fused conv3x3 + tanh + relu + 1x1 + sigmoid via bf16 WMMA
    conv_wmma_kernel<<<dim3(MBLOCKS, BATCH), 256, 0, stream>>>(x, b_rnn, w_out, b_out,
                                                               Bp, score);

    // 3) per-batch argmax position
    argmax_kernel<<<BATCH, 256, 0, stream>>>(score, pos);
}
